// FourierInterpolator_39771397161482
// MI455X (gfx1250) — matrix operand order repaired and checked by reading, not compiled, not run on hardware
//
#include <hip/hip_runtime.h>

typedef __attribute__((ext_vector_type(16))) _Float16 v16h;
typedef __attribute__((ext_vector_type(8)))  _Float16 v8h;
typedef __attribute__((ext_vector_type(8)))  float    v8f;

#define TWO_PI_F 6.28318530717958647692f

// ---------------------------------------------------------------------------
// Kernel 1 (prep): per batch, direct 32x32 DFT of y, then build the 64x64
// bilinear-form matrix U = [[P Q],[Q R]] (f16, scaled by 1/1024) in workspace.
//   P =  re^2 / |c| / 1024,  Q = -re*im / |c| / 1024,  R = im^2 / |c| / 1024
// One 1024-thread block per batch; thread = one (k1,k2) frequency pair.
// ---------------------------------------------------------------------------
__global__ __launch_bounds__(1024)
void fourier_prep(const float* __restrict__ y, _Float16* __restrict__ Uws)
{
    __shared__ float ly[1024];
    __shared__ float zr[1024];
    __shared__ float zi[1024];

    const int b = blockIdx.x;
    const int t = threadIdx.x;

    ly[t] = y[b * 1024 + t];
    __syncthreads();

    // stage 1: DFT along n2; thread = (n1, k2)
    {
        const int n1 = t >> 5, k2 = t & 31;
        const float w = (TWO_PI_F / 32.0f) * (float)k2;
        float ar = 0.0f, ai = 0.0f;
        for (int n2 = 0; n2 < 32; ++n2) {
            float s, c;
            __sincosf(w * (float)n2, &s, &c);
            const float v = ly[n1 * 32 + n2];
            ar += v * c;
            ai -= v * s;
        }
        zr[t] = ar;
        zi[t] = ai;
    }
    __syncthreads();

    // stage 2: DFT along n1; thread = (k1, k2)
    const int k1 = t >> 5, k2 = t & 31;
    const float w = (TWO_PI_F / 32.0f) * (float)k1;
    float re = 0.0f, im = 0.0f;
    for (int n1 = 0; n1 < 32; ++n1) {
        float s, c;
        __sincosf(w * (float)n1, &s, &c);
        const float a  = zr[n1 * 32 + k2];
        const float bb = zi[n1 * 32 + k2];
        re += a * c + bb * s;
        im += bb * c - a * s;
    }

    const float mag2 = re * re + im * im;
    float inv = 0.0f;
    if (mag2 > 1e-24f) inv = 1.0f / (1024.0f * sqrtf(mag2));
    const float P =  re * re * inv;
    const float Q = -re * im * inv;
    const float R =  im * im * inv;

    _Float16* U = Uws + b * 4096;            // 64x64 row-major per batch
    U[k1 * 64 + k2]             = (_Float16)P;
    U[k1 * 64 + 32 + k2]        = (_Float16)Q;
    U[(32 + k1) * 64 + k2]      = (_Float16)Q;
    U[(32 + k1) * 64 + 32 + k2] = (_Float16)R;
}

// ---------------------------------------------------------------------------
// Kernel 2 (eval): each wave32 owns one batch b and a tile of 16 samples.
//   W(64x16) = U(64x64) @ V(64x16)  via 8x v_wmma_f32_16x16x32_f16
//   out[m]   = a(m) . W[:,m]        (per-lane dot + cross-half shuffle reduce)
// V columns are [cos(2*pi*x1*k); sin(2*pi*x1*k)], a = same with x0.
// 2048 wave-tiles total -> 256 blocks x 8 waves.
// ---------------------------------------------------------------------------
__global__ __launch_bounds__(256)
void fourier_eval(const _Float16* __restrict__ Uws,
                  const float* __restrict__ xnew,
                  float* __restrict__ out)
{
    const int lane  = threadIdx.x & 31;
    const int wave  = threadIdx.x >> 5;
    const int tile  = blockIdx.x * 8 + wave;   // [0, 2048)
    const int b     = tile >> 8;               // 256 tiles per batch
    const int mtile = (tile & 255) << 4;       // 16 samples per wave-tile

    const _Float16* U = Uws + b * 4096;
    __builtin_prefetch(U, 0, 0);               // global_prefetch_b8 on U block

    const int n    = lane & 15;                // sample slot / matrix column
    const int half = lane >> 4;                // 0: lanes 0-15, 1: lanes 16-31
    const int m    = mtile + n;
    const float x0 = xnew[(b * 4096 + m) * 2 + 0];
    const float x1 = xnew[(b * 4096 + m) * 2 + 1];

    // B operands (same K octets as A: lanes 0-15 hold K=0..7,16..23, lanes 16-31 K=8..15,24..31):
    //   K-step 0 -> cos(2*pi*x1*k), K-step 1 -> sin(2*pi*x1*k)
    const float th2  = TWO_PI_F * x1;
    const int  kbase = half * 8;
    v16h Bc, Bs;
#pragma unroll
    for (int tt = 0; tt < 16; ++tt) {
        float s, c;
        __sincosf(th2 * (float)(kbase + tt + (tt & 8)), &s, &c);
        Bc[tt] = (_Float16)c;
        Bs[tt] = (_Float16)s;
    }

    // A layout: lanes 0-15 hold K in {k0..k0+7, k0+16..k0+23} of row 16r+n,
    //           lanes 16-31 the complementary K octets (off = 8).
    const int off = half * 8;
    v8f acc0 = {}, acc1 = {}, acc2 = {}, acc3 = {};

#pragma unroll
    for (int s = 0; s < 2; ++s) {              // K-steps over U columns
        const int k0 = s * 32;
        const v16h Bop = s ? Bs : Bc;
#pragma unroll
        for (int r = 0; r < 4; ++r) {          // 4 row-tiles of W
            const int row = r * 16 + n;
            const _Float16* rp = U + row * 64 + k0 + off;
            const v8h lo = *(const v8h*)rp;
            const v8h hi = *(const v8h*)(rp + 16);
            v16h A;
#pragma unroll
            for (int i = 0; i < 8; ++i) { A[i] = lo[i]; A[8 + i] = hi[i]; }
            v8f& acc = (r == 0) ? acc0 : (r == 1) ? acc1 : (r == 2) ? acc2 : acc3;
            acc = __builtin_amdgcn_wmma_f32_16x16x32_f16(
                      false, A, false, Bop, (short)0, acc, false, false);
        }
    }

    // Epilogue: out[m] = sum_i a_i(m) * W[i][m]; lane holds rows
    //   acc0: i = off+v, acc1: i = 16+off+v (cos rows)
    //   acc2: i = 32+off+v, acc3: i = 48+off+v (sin rows)
    const float th1 = TWO_PI_F * x0;
    float partial = 0.0f;
#pragma unroll
    for (int v = 0; v < 8; ++v) {
        float s0, c0, s1, c1;
        __sincosf(th1 * (float)(off + v),      &s0, &c0);
        __sincosf(th1 * (float)(16 + off + v), &s1, &c1);
        partial += c0 * acc0[v] + c1 * acc1[v] + s0 * acc2[v] + s1 * acc3[v];
    }
    partial += __shfl_xor(partial, 16, 32);    // combine the two half-wave row sets

    if (half == 0) out[b * 4096 + mtile + n] = partial;
}

// ---------------------------------------------------------------------------
extern "C" void kernel_launch(void* const* d_in, const int* in_sizes, int n_in,
                              void* d_out, int out_size, void* d_ws, size_t ws_size,
                              hipStream_t stream)
{
    (void)in_sizes; (void)n_in; (void)out_size; (void)ws_size;
    const float* y    = (const float*)d_in[0];   // (8, 32, 32) f32
    const float* xnew = (const float*)d_in[1];   // (8, 4096, 2) f32
    float*       out  = (float*)d_out;           // (8, 64, 64) f32
    _Float16*    Uws  = (_Float16*)d_ws;         // 8 * 64*64 f16 = 64 KB

    fourier_prep<<<8, 1024, 0, stream>>>(y, Uws);
    fourier_eval<<<256, 256, 0, stream>>>(Uws, xnew, out);
}
